// GraphNormalizingFlow_58875411693667
// MI455X (gfx1250) — compile-verified, matrix operand order checked by reading, not run on hardware
//
#include <hip/hip_runtime.h>
#include <hip/hip_bf16.h>
#include <cstdint>

typedef __bf16 v16bf __attribute__((ext_vector_type(16)));
typedef float  v8f   __attribute__((ext_vector_type(8)));

#define B_ROWS  16384
#define D_DIM   3876          // 38*64 + 38*38
#define CONDIM  1938          // (D+1)/2 ; D - CONDIM = 1938
#define HALFC   969           // CONDIM/2
#define HIDDIM  1024
#define NLAYERS 8
#define Z1P     1952          // packed z1 row stride (bf16), 16B-aligned rows

constexpr int BM  = 128;      // rows per block
constexpr int BN  = 64;       // cols per block
constexpr int BK  = 32;       // k per step (matches wmma 16x16x32 bf16)
constexpr int LDA = BK + 16;  // 48 bf16 = 96B row stride -> 32B-aligned frags
constexpr int LDB = BK + 16;

template <bool V> struct BoolC { static constexpr bool value = V; };

// 32B raw copy global(bf16) -> LDS via CDNA5 async-to-LDS (ASYNCcnt-tracked).
__device__ __forceinline__ void async_copy_32B(const __bf16* gsrc, __bf16* lds_dst)
{
    const unsigned           loff = (unsigned)(uintptr_t)lds_dst;
    const unsigned long long ga   = (unsigned long long)(uintptr_t)gsrc;
    asm volatile("global_load_async_to_lds_b128 %0, %1, off\n\t"
                 "global_load_async_to_lds_b128 %0, %1, off offset:16"
                 :: "v"(loff), "v"(ga) : "memory");
}
__device__ __forceinline__ void async_wait0()
{
    asm volatile("s_wait_asynccnt 0x0" ::: "memory");
}

// ---------------------------------------------------------------------------
// Prepass: pack even columns of x into bf16 z1 buffer (layer 0 operand).
// ---------------------------------------------------------------------------
__global__ __launch_bounds__(256)
void flow_pack_x(const float* __restrict__ x, __bf16* __restrict__ z1)
{
    const int row = blockIdx.x;
    const long  gb = (long)row * D_DIM;
    const long  pb = (long)row * Z1P;
    for (int c = threadIdx.x; c < CONDIM; c += 256)
        z1[pb + c] = (__bf16)x[gb + 2 * c];
}

// ---------------------------------------------------------------------------
// GEMM1 + ReLU: h = relu(z1 @ W1 + b1); A from packed bf16 z1 (async->LDS).
// n-block 0 also: fp32 z1 passthrough into dstz[:, :CONDIM] and packs the
// even-even entries into z1nxt[:, 0:HALFC].
// ---------------------------------------------------------------------------
__global__ __launch_bounds__(256)
void flow_gemm1_relu(const float*  __restrict__ src,    // previous z, fp32
                     const __bf16* __restrict__ z1cur,  // packed bf16 z1
                     const float*  __restrict__ W1l,    // (CONDIM, HIDDIM)
                     const float*  __restrict__ b1l,    // (HIDDIM)
                     __bf16*       __restrict__ hbuf,   // (B_ROWS, HIDDIM) bf16
                     float*        __restrict__ dstz,   // (B_ROWS, D_DIM)
                     __bf16*       __restrict__ z1nxt)  // packed z1 for layer i+1
{
    __shared__ alignas(32) __bf16 lds_a[BM * LDA];
    __shared__ alignas(32) __bf16 lds_b[BN * LDB];

    const int tid   = threadIdx.x;
    const int lane  = tid & 31;
    const int wave  = tid >> 5;
    const int wm    = wave & 3;   // 4 m-slices of 32 rows
    const int wn    = wave >> 2;  // 2 n-slices of 32 cols
    const int lmod  = lane & 15;
    const int lhalf = lane >> 4;

    const int n0 = blockIdx.x * BN;
    const int m0 = blockIdx.y * BM;
    const bool doCopy = (blockIdx.x == 0);

    v8f acc[2][2] = {};

    const int  a_row   = tid >> 1;
    const int  a_kh    = (tid & 1) * 16;
    const long a_gbase = (long)(m0 + a_row) * D_DIM;
    const long a_pbase = (long)(m0 + a_row) * Z1P;
    const int  b_n  = tid & 63;
    const int  b_kb = (tid >> 6) * 8;

    auto tile_step = [&](int k0, auto tail_c) {
        constexpr bool TAIL = decltype(tail_c)::value;
        __syncthreads();

        // ---- A tile: raw 32B async copy from packed bf16 z1
        async_copy_32B(z1cur + a_pbase + k0 + a_kh, &lds_a[a_row * LDA + a_kh]);

        // ---- fp32 passthrough + z1nxt packing (n-block 0 only)
        if (doCopy) {
            if (!TAIL) {
                float va[16];
                #pragma unroll
                for (int q = 0; q < 8; ++q) {
                    const int kg = k0 + a_kh + 2 * q;
                    const float4 p = *(const float4*)&src[a_gbase + 2 * kg];
                    va[2 * q] = p.x; va[2 * q + 1] = p.z;
                }
                #pragma unroll
                for (int q = 0; q < 4; ++q)
                    *(float4*)&dstz[a_gbase + k0 + a_kh + 4 * q] =
                        make_float4(va[4*q], va[4*q+1], va[4*q+2], va[4*q+3]);
                alignas(16) __bf16 pe[8];
                #pragma unroll
                for (int e = 0; e < 8; ++e) pe[e] = (__bf16)va[2 * e];  // even kg
                *(uint4*)&z1nxt[a_pbase + (k0 + a_kh) / 2] = *(const uint4*)pe;
            } else {
                #pragma unroll
                for (int kk = 0; kk < 16; ++kk) {
                    const int kg = k0 + a_kh + kk;
                    if (kg < CONDIM) {
                        const float v = src[a_gbase + 2 * kg];
                        dstz[a_gbase + kg] = v;
                        if ((kg & 1) == 0) z1nxt[a_pbase + kg / 2] = (__bf16)v;
                    }
                }
            }
        }

        // ---- B tile: one n column, 8 k values -> single b128 store
        {
            alignas(16) __bf16 bb[8];
            #pragma unroll
            for (int j = 0; j < 8; ++j) {
                const int kg = k0 + b_kb + j;
                if (!TAIL) {
                    bb[j] = (__bf16)W1l[(long)kg * HIDDIM + n0 + b_n];
                } else {
                    const int kc = kg < CONDIM ? kg : CONDIM - 1;
                    const float v = W1l[(long)kc * HIDDIM + n0 + b_n];
                    bb[j] = (__bf16)((kg < CONDIM) ? v : 0.0f);  // zero pad rows
                }
            }
            *(uint4*)&lds_b[b_n * LDB + b_kb] = *(const uint4*)bb;
        }
        if (!TAIL) {
            __builtin_prefetch(z1cur + a_pbase + k0 + BK + a_kh, 0, 3);
            __builtin_prefetch(&W1l[(long)(k0 + BK + b_kb) * HIDDIM + n0 + b_n], 0, 3);
        }
        async_wait0();
        __syncthreads();

        v16bf af[2], bf[2];
        #pragma unroll
        for (int mi = 0; mi < 2; ++mi)
            af[mi] = *(const v16bf*)&lds_a[(wm * 32 + mi * 16 + lmod) * LDA + lhalf * 16];
        #pragma unroll
        for (int ni = 0; ni < 2; ++ni)
            bf[ni] = *(const v16bf*)&lds_b[(wn * 32 + ni * 16 + lmod) * LDB + lhalf * 16];
        #pragma unroll
        for (int mi = 0; mi < 2; ++mi)
            #pragma unroll
            for (int ni = 0; ni < 2; ++ni)
                acc[mi][ni] = __builtin_amdgcn_wmma_f32_16x16x32_bf16(
                    false, af[mi], false, bf[ni], (short)0, acc[mi][ni], false, false);
    };

    constexpr int KT_FULL = CONDIM / BK;          // 60 full tiles
    for (int kt = 0; kt < KT_FULL; ++kt) tile_step(kt * BK, BoolC<false>{});
    tile_step(KT_FULL * BK, BoolC<true>{});       // tail: B rows zeroed

    // ---- epilogue: relu(acc + bias) -> bf16 h
    #pragma unroll
    for (int ni = 0; ni < 2; ++ni) {
        const int col  = n0 + wn * 32 + ni * 16 + lmod;
        const float bv = b1l[col];
        #pragma unroll
        for (int mi = 0; mi < 2; ++mi) {
            #pragma unroll
            for (int r = 0; r < 8; ++r) {
                const int row = m0 + wm * 32 + mi * 16 + lhalf * 8 + r;
                float v = acc[mi][ni][r] + bv;
                v = v > 0.0f ? v : 0.0f;
                hbuf[(long)row * HIDDIM + col] = (__bf16)v;
            }
        }
    }
}

// ---------------------------------------------------------------------------
// GEMM2 + coupling: st = h @ W2 + b2 ; z2_new = z2*exp(-s)+t ;
// logdet -= rowsum(s) ; packs even-o z2_new into z1nxt[:, HALFC:].
// ---------------------------------------------------------------------------
__global__ __launch_bounds__(256)
void flow_gemm2_couple(const float*  __restrict__ src,    // previous z (for z2)
                       const __bf16* __restrict__ hbuf,
                       const float*  __restrict__ W2l,    // (HIDDIM, 2*CONDIM)
                       const float*  __restrict__ b2l,    // (2*CONDIM)
                       float*        __restrict__ dstz,
                       float*        __restrict__ logdet,
                       __bf16*       __restrict__ z1nxt)
{
    __shared__ alignas(32) __bf16 lds_a [BM * LDA];
    __shared__ alignas(32) __bf16 lds_bs[BN * LDB];
    __shared__ alignas(32) __bf16 lds_bt[BN * LDB];
    __shared__ float lds_rowsum[BM];

    const int tid   = threadIdx.x;
    const int lane  = tid & 31;
    const int wave  = tid >> 5;
    const int wm    = wave & 3;
    const int wn    = wave >> 2;
    const int lmod  = lane & 15;
    const int lhalf = lane >> 4;

    const int n0 = blockIdx.x * BN;
    const int m0 = blockIdx.y * BM;

    if (tid < BM) lds_rowsum[tid] = 0.0f;

    v8f acc_s[2][2] = {};
    v8f acc_t[2][2] = {};

    const int a_row = tid >> 1;
    const int a_kh  = (tid & 1) * 16;
    const int b_n   = tid & 63;
    const int b_kb  = (tid >> 6) * 8;
    const int ng    = n0 + b_n;
    const bool nok  = ng < CONDIM;    // per-thread-uniform column guard

    const int KT = HIDDIM / BK;   // 32, exact
    for (int kt = 0; kt < KT; ++kt) {
        const int k0 = kt * BK;
        __syncthreads();

        // ---- A tile: raw 32B async copy from bf16 h
        async_copy_32B(hbuf + (long)(m0 + a_row) * HIDDIM + k0 + a_kh,
                       &lds_a[a_row * LDA + a_kh]);

        // ---- stage s/t halves of W2: one n column, 8 k -> single b128 stores
        {
            alignas(16) __bf16 vs8[8], vt8[8];
            #pragma unroll
            for (int j = 0; j < 8; ++j) {
                const long wrow = (long)(k0 + b_kb + j) * D_DIM;
                float vs = 0.0f, vt = 0.0f;
                if (nok) { vs = W2l[wrow + ng]; vt = W2l[wrow + CONDIM + ng]; }
                vs8[j] = (__bf16)vs;
                vt8[j] = (__bf16)vt;
            }
            *(uint4*)&lds_bs[b_n * LDB + b_kb] = *(const uint4*)vs8;
            *(uint4*)&lds_bt[b_n * LDB + b_kb] = *(const uint4*)vt8;
        }
        if (kt + 1 < KT) {
            __builtin_prefetch(hbuf + (long)(m0 + a_row) * HIDDIM + k0 + BK + a_kh, 0, 3);
            __builtin_prefetch(&W2l[(long)(k0 + BK + b_kb) * D_DIM + ng], 0, 3);
        }
        async_wait0();
        __syncthreads();

        v16bf af[2], bsf[2], btf[2];
        #pragma unroll
        for (int mi = 0; mi < 2; ++mi)
            af[mi] = *(const v16bf*)&lds_a[(wm * 32 + mi * 16 + lmod) * LDA + lhalf * 16];
        #pragma unroll
        for (int ni = 0; ni < 2; ++ni) {
            bsf[ni] = *(const v16bf*)&lds_bs[(wn * 32 + ni * 16 + lmod) * LDB + lhalf * 16];
            btf[ni] = *(const v16bf*)&lds_bt[(wn * 32 + ni * 16 + lmod) * LDB + lhalf * 16];
        }
        #pragma unroll
        for (int mi = 0; mi < 2; ++mi)
            #pragma unroll
            for (int ni = 0; ni < 2; ++ni) {
                acc_s[mi][ni] = __builtin_amdgcn_wmma_f32_16x16x32_bf16(
                    false, af[mi], false, bsf[ni], (short)0, acc_s[mi][ni], false, false);
                acc_t[mi][ni] = __builtin_amdgcn_wmma_f32_16x16x32_bf16(
                    false, af[mi], false, btf[ni], (short)0, acc_t[mi][ni], false, false);
            }
    }

    // ---- fused coupling epilogue
    #pragma unroll
    for (int ni = 0; ni < 2; ++ni) {
        const int col = n0 + wn * 32 + ni * 16 + lmod;   // o index into s/t
        if (col < CONDIM) {
            const float bs = b2l[col];
            const float bt = b2l[CONDIM + col];
            #pragma unroll
            for (int mi = 0; mi < 2; ++mi) {
                #pragma unroll
                for (int r = 0; r < 8; ++r) {
                    const int rloc = wm * 32 + mi * 16 + lhalf * 8 + r;
                    const int row  = m0 + rloc;
                    const float s  = acc_s[mi][ni][r] + bs;
                    const float t  = acc_t[mi][ni][r] + bt;
                    const float z2 = src[(long)row * D_DIM + 2 * col + 1];
                    const float zn = z2 * __expf(-s) + t;
                    dstz[(long)row * D_DIM + CONDIM + col] = zn;
                    if ((col & 1) == 0)   // even o -> next layer's z1, packed bf16
                        z1nxt[(long)row * Z1P + HALFC + col / 2] = (__bf16)zn;
                    atomicAdd(&lds_rowsum[rloc], s);          // ds_add_f32
                }
            }
        }
    }
    __syncthreads();
    if (tid < BM) {
        const float v = lds_rowsum[tid];
        atomicAdd(&logdet[m0 + tid], -v);                     // global f32 atomic
    }
}

__global__ void flow_init_logdet(float* __restrict__ logdet)
{
    const int i = blockIdx.x * blockDim.x + threadIdx.x;
    if (i < B_ROWS) logdet[i] = 0.0f;
}

// ---------------------------------------------------------------------------
extern "C" void kernel_launch(void* const* d_in, const int* in_sizes, int n_in,
                              void* d_out, int out_size, void* d_ws, size_t ws_size,
                              hipStream_t stream)
{
    (void)in_sizes; (void)n_in; (void)out_size; (void)ws_size;

    const float* x  = (const float*)d_in[0];
    const float* W1 = (const float*)d_in[1];  // (8, CONDIM, HIDDIM)
    const float* b1 = (const float*)d_in[2];  // (8, HIDDIM)
    const float* W2 = (const float*)d_in[3];  // (8, HIDDIM, 2*CONDIM)
    const float* b2 = (const float*)d_in[4];  // (8, 2*CONDIM)

    float* out_z  = (float*)d_out;
    float* logdet = out_z + (size_t)B_ROWS * D_DIM;

    // workspace layout: fp32 z ping | bf16 h | bf16 z1 ping A | bf16 z1 ping B
    char*  p    = (char*)d_ws;
    float* ws_z = (float*)p;
    p += (((size_t)B_ROWS * D_DIM * sizeof(float)) + 255) & ~(size_t)255;
    __bf16* hbuf = (__bf16*)p;
    p += (((size_t)B_ROWS * HIDDIM * sizeof(__bf16)) + 255) & ~(size_t)255;
    __bf16* z1a = (__bf16*)p;
    p += (((size_t)B_ROWS * Z1P * sizeof(__bf16)) + 255) & ~(size_t)255;
    __bf16* z1b = (__bf16*)p;
    __bf16* z1bufs[2] = { z1a, z1b };

    flow_init_logdet<<<(B_ROWS + 255) / 256, 256, 0, stream>>>(logdet);
    flow_pack_x<<<B_ROWS, 256, 0, stream>>>(x, z1a);

    const dim3 blk(256);
    const dim3 g1(HIDDIM / BN, B_ROWS / BM);                 // 16 x 128
    const dim3 g2((CONDIM + BN - 1) / BN, B_ROWS / BM);      // 31 x 128

    const float* src = x;
    for (int i = 0; i < NLAYERS; ++i) {
        float* dst = (i & 1) ? out_z : ws_z;   // layer 7 lands in out_z
        __bf16* z1cur = z1bufs[i & 1];
        __bf16* z1nxt = z1bufs[(i + 1) & 1];
        const float* W1l = W1 + (size_t)i * CONDIM * HIDDIM;
        const float* b1l = b1 + (size_t)i * HIDDIM;
        const float* W2l = W2 + (size_t)i * HIDDIM * (2 * CONDIM);
        const float* b2l = b2 + (size_t)i * (2 * CONDIM);

        flow_gemm1_relu<<<g1, blk, 0, stream>>>(src, z1cur, W1l, b1l, hbuf, dst, z1nxt);
        flow_gemm2_couple<<<g2, blk, 0, stream>>>(src, hbuf, W2l, b2l, dst, logdet, z1nxt);

        src = dst;
    }
}
